// InteractionAwarePredictor_23776938951051
// MI455X (gfx1250) — compile-verified, matrix operand order checked by reading, not run on hardware
//
#include <hip/hip_runtime.h>
#include <math.h>

typedef __attribute__((ext_vector_type(16))) _Float16 v16h;
typedef __attribute__((ext_vector_type(8)))  float    v8f;

// ---------------------------------------------------------------- helpers

__device__ __forceinline__ v8f vzero8() {
  v8f v;
#pragma unroll
  for (int i = 0; i < 8; ++i) v[i] = 0.0f;
  return v;
}

__device__ __forceinline__ v8f wmma_f16(v16h a, v16h b, v8f c) {
  // D = A(16x32 f16) * B(32x16 f16) + C(16x16 f32)
  return __builtin_amdgcn_wmma_f32_16x16x32_f16(false, a, false, b, (short)0, c,
                                                false, false);
}

// A-matrix (16x32) lane layout loader from row-major f32 (stride floats/row).
// lane: M = lane&15, half = lane>>4. element e<8 -> K = 8*half+e,
// e>=8 -> K = 16 + 8*half + (e-8).
__device__ __forceinline__ v16h load_a_f32(const float* base, int stride, int lane) {
  const int m = lane & 15, half = (lane >> 4) & 1;
  const float* r = base + m * stride;
  v16h a;
#pragma unroll
  for (int e = 0; e < 8; ++e) a[e] = (_Float16)r[8 * half + e];
#pragma unroll
  for (int e = 0; e < 8; ++e) a[8 + e] = (_Float16)r[16 + 8 * half + e];
  return a;
}

// A-matrix loader where rows are 16 floats (K=16..31 are zero)
__device__ __forceinline__ v16h load_a_pad16(const float* base, int lane) {
  const int m = lane & 15, half = (lane >> 4) & 1;
  const float* r = base + m * 16;
  v16h a;
#pragma unroll
  for (int e = 0; e < 8; ++e) a[e] = (_Float16)r[8 * half + e];
#pragma unroll
  for (int e = 0; e < 8; ++e) a[8 + e] = (_Float16)0.0f;
  return a;
}

// packed B/A tile: 32 lanes x 16 halves, contiguous per tile (512 halves)
__device__ __forceinline__ v16h load_packed(const _Float16* p, int tile, int lane) {
  return ((const v16h*)p)[tile * 32 + lane];
}

__device__ __forceinline__ float sigmoidf_(float x) { return 1.0f / (1.0f + expf(-x)); }

// ---------------------------------------------------------------- utility kernels

__global__ void fill_kernel(float* p, float v, long n) {
  long i = (long)blockIdx.x * blockDim.x + threadIdx.x;
  if (i < n) p[i] = v;
}

__global__ void relu_kernel(const float* in, float* out, long n) {
  long i = (long)blockIdx.x * blockDim.x + threadIdx.x;
  if (i < n) out[i] = fmaxf(in[i], 0.0f);
}

__global__ void deg_kernel(const int* dst, float* deg, int E) {
  int i = blockIdx.x * blockDim.x + threadIdx.x;
  if (i < E) atomicAdd(&deg[dst[i]], 1.0f);
}

__global__ void node_norm_kernel(const float* deg, float* dinv, float* snorm, int N) {
  int i = blockIdx.x * blockDim.x + threadIdx.x;
  if (i < N) {
    float d = deg[i] + 1.0f;
    dinv[i] = rsqrtf(d);
    snorm[i] = 1.0f / d;
  }
}

__global__ void enorm_kernel(const int* src, const int* dst, const float* dinv,
                             float* enorm, int E) {
  int i = blockIdx.x * blockDim.x + threadIdx.x;
  if (i < E) enorm[i] = dinv[src[i]] * dinv[dst[i]];
}

__global__ void softmax30_kernel(const float* att, float* probs) {
  int i = threadIdx.x;
  float v = (i < 30) ? att[i] : -1e30f;
  float mx = v;
#pragma unroll
  for (int off = 16; off > 0; off >>= 1) mx = fmaxf(mx, __shfl_xor(mx, off));
  float e = (i < 30) ? expf(v - mx) : 0.0f;
  float s = e;
#pragma unroll
  for (int off = 16; off > 0; off >>= 1) s += __shfl_xor(s, off);
  if (i < 30) probs[i] = e / s;
}

// Pack f32 weight matrix (Kin x Nin row-major) into WMMA B-tile layout.
// grid.x = kchunks*ntiles; tile = kc*ntiles + nt.
__global__ void pack_b_kernel(const float* W, int Kin, int Nin, int ntiles,
                              _Float16* out) {
  int tile = blockIdx.x, lane = threadIdx.x;
  int kc = tile / ntiles, nt = tile - kc * ntiles;
  int n = (lane & 15) + nt * 16;
  int half = lane >> 4;
#pragma unroll
  for (int e = 0; e < 16; ++e) {
    int j = e >> 1;
    int k = kc * 32 + ((j >> 2) << 4) + ((j & 3) << 1) + (half << 3) + (e & 1);
    float v = (k < Kin && n < Nin) ? W[k * Nin + n] : 0.0f;
    out[tile * 512 + lane * 16 + e] = (_Float16)v;
  }
}

// ---------------------------------------------------------------- aggregation

// layer1: aggX[n][0..8] = snorm[n]*x[n,f,t]   (cols 9..15 = 0), then edge atomics
__global__ void aggx_init_kernel(const float* x, const float* snorm, float* aggX,
                                 int t, int N) {
  int i = blockIdx.x * blockDim.x + threadIdx.x;
  if (i >= N * 16) return;
  int n = i >> 4, f = i & 15;
  aggX[i] = (f < 9) ? snorm[n] * x[(n * 9 + f) * 30 + t] : 0.0f;
}

__global__ void aggx_edge_kernel(const float* x, const int* src, const int* dst,
                                 const float* enorm, float* aggX, int t, int E) {
  int i = blockIdx.x * blockDim.x + threadIdx.x;
  if (i >= E * 9) return;
  int e = i / 9, f = i - e * 9;
  float v = enorm[e] * x[(src[e] * 9 + f) * 30 + t];
  atomicAdd(&aggX[dst[e] * 16 + f], v);
}

// layer2: 32-wide aggregation of hL1 (done once)
__global__ void aggh_init_kernel(const float* hL1, const float* snorm, float* aggH,
                                 int N) {
  int i = blockIdx.x * blockDim.x + threadIdx.x;
  if (i < N * 32) aggH[i] = snorm[i >> 5] * hL1[i];
}

__global__ void aggh_edge_kernel(const float* hL1, const int* src, const int* dst,
                                 const float* enorm, float* aggH, int E) {
  int i = blockIdx.x * blockDim.x + threadIdx.x;
  if (i >= E * 32) return;
  int e = i >> 5, k = i & 31;
  atomicAdd(&aggH[dst[e] * 32 + k], enorm[e] * hL1[src[e] * 32 + k]);
}

// ---------------------------------------------------------------- GRU cells

// Layer-1 fused GRU cell: one wave per 16-node tile.
__global__ __launch_bounds__(32)
void l1_cell_kernel(const float* __restrict__ aggX, float* __restrict__ H,
                    float* __restrict__ acc,
                    const _Float16* __restrict__ pWg,   // [3 gates][2 ntiles]
                    const _Float16* __restrict__ pWl,   // [3][2 kchunk][2 ntile]
                    const float* bg_z, const float* bg_r, const float* bg_h,
                    const float* bl_z, const float* bl_r, const float* bl_h,
                    const float* probs, int t) {
  __shared__ float lds[16 * 32];
  const int tile = blockIdx.x, lane = threadIdx.x;
  const int m = lane & 15, half = lane >> 4;
  const int base = tile * 16;

  v16h aX = load_a_pad16(aggX + (long)base * 16, lane);
  v16h aH = load_a_f32(H + (long)base * 32, 32, lane);

  v8f Hd0, Hd1;  // H in D layout, two 16-col tiles
#pragma unroll
  for (int v = 0; v < 8; ++v) {
    int row = base + v + 8 * half;
    Hd0[v] = H[row * 32 + m];
    Hd1[v] = H[row * 32 + 16 + m];
  }

  const float* bgp[3] = {bg_z, bg_r, bg_h};
  v16h aG[3];  // GCN outputs per gate, A layout
#pragma unroll
  for (int g = 0; g < 3; ++g) {
    v8f d0 = vzero8(), d1 = vzero8();
    d0 = wmma_f16(aX, load_packed(pWg, g * 2 + 0, lane), d0);
    d1 = wmma_f16(aX, load_packed(pWg, g * 2 + 1, lane), d1);
    float b0 = bgp[g][m], b1 = bgp[g][16 + m];
#pragma unroll
    for (int v = 0; v < 8; ++v) {
      lds[(v + 8 * half) * 32 + m] = d0[v] + b0;
      lds[(v + 8 * half) * 32 + 16 + m] = d1[v] + b1;
    }
    __syncthreads();
    aG[g] = load_a_f32(lds, 32, lane);
    __syncthreads();
  }

  // z gate
  v8f zd0 = vzero8(), zd1 = vzero8();
  zd0 = wmma_f16(aG[0], load_packed(pWl, 0 * 4 + 0 * 2 + 0, lane), zd0);
  zd0 = wmma_f16(aH, load_packed(pWl, 0 * 4 + 1 * 2 + 0, lane), zd0);
  zd1 = wmma_f16(aG[0], load_packed(pWl, 0 * 4 + 0 * 2 + 1, lane), zd1);
  zd1 = wmma_f16(aH, load_packed(pWl, 0 * 4 + 1 * 2 + 1, lane), zd1);
  {
    float b0 = bl_z[m], b1 = bl_z[16 + m];
#pragma unroll
    for (int v = 0; v < 8; ++v) {
      zd0[v] = sigmoidf_(zd0[v] + b0);
      zd1[v] = sigmoidf_(zd1[v] + b1);
    }
  }
  // r gate
  v8f rd0 = vzero8(), rd1 = vzero8();
  rd0 = wmma_f16(aG[1], load_packed(pWl, 1 * 4 + 0 * 2 + 0, lane), rd0);
  rd0 = wmma_f16(aH, load_packed(pWl, 1 * 4 + 1 * 2 + 0, lane), rd0);
  rd1 = wmma_f16(aG[1], load_packed(pWl, 1 * 4 + 0 * 2 + 1, lane), rd1);
  rd1 = wmma_f16(aH, load_packed(pWl, 1 * 4 + 1 * 2 + 1, lane), rd1);
  {
    float b0 = bl_r[m], b1 = bl_r[16 + m];
#pragma unroll
    for (int v = 0; v < 8; ++v) {
      rd0[v] = sigmoidf_(rd0[v] + b0);
      rd1[v] = sigmoidf_(rd1[v] + b1);
    }
  }
  // H*R -> A layout via LDS
#pragma unroll
  for (int v = 0; v < 8; ++v) {
    lds[(v + 8 * half) * 32 + m] = Hd0[v] * rd0[v];
    lds[(v + 8 * half) * 32 + 16 + m] = Hd1[v] * rd1[v];
  }
  __syncthreads();
  v16h aHR = load_a_f32(lds, 32, lane);
  __syncthreads();
  // h gate
  v8f hd0 = vzero8(), hd1 = vzero8();
  hd0 = wmma_f16(aG[2], load_packed(pWl, 2 * 4 + 0 * 2 + 0, lane), hd0);
  hd0 = wmma_f16(aHR, load_packed(pWl, 2 * 4 + 1 * 2 + 0, lane), hd0);
  hd1 = wmma_f16(aG[2], load_packed(pWl, 2 * 4 + 0 * 2 + 1, lane), hd1);
  hd1 = wmma_f16(aHR, load_packed(pWl, 2 * 4 + 1 * 2 + 1, lane), hd1);
  {
    float b0 = bl_h[m], b1 = bl_h[16 + m];
#pragma unroll
    for (int v = 0; v < 8; ++v) {
      hd0[v] = tanhf(hd0[v] + b0);
      hd1[v] = tanhf(hd1[v] + b1);
    }
  }
  // H = z*H + (1-z)*ht ; acc += w*H
  float w = probs[t];
#pragma unroll
  for (int v = 0; v < 8; ++v) {
    int row = base + v + 8 * half;
    float hn0 = zd0[v] * Hd0[v] + (1.0f - zd0[v]) * hd0[v];
    float hn1 = zd1[v] * Hd1[v] + (1.0f - zd1[v]) * hd1[v];
    H[row * 32 + m] = hn0;
    H[row * 32 + 16 + m] = hn1;
    acc[row * 32 + m] += w * hn0;
    acc[row * 32 + 16 + m] += w * hn1;
  }
}

// Layer-2 GCN outputs (period invariant): compute once into packed f16 A-tiles.
__global__ __launch_bounds__(32)
void l2_gcn_kernel(const float* __restrict__ aggH,
                   const _Float16* __restrict__ pWg,  // [3][2]
                   const float* bg_z, const float* bg_r, const float* bg_h,
                   _Float16* gAz, _Float16* gAr, _Float16* gAh) {
  __shared__ float lds[16 * 32];
  const int tile = blockIdx.x, lane = threadIdx.x;
  const int m = lane & 15, half = lane >> 4;
  const int base = tile * 16;
  v16h aA = load_a_f32(aggH + (long)base * 32, 32, lane);
  const float* bgp[3] = {bg_z, bg_r, bg_h};
  _Float16* outp[3] = {gAz, gAr, gAh};
#pragma unroll
  for (int g = 0; g < 3; ++g) {
    v8f d0 = vzero8(), d1 = vzero8();
    d0 = wmma_f16(aA, load_packed(pWg, g * 2 + 0, lane), d0);
    d1 = wmma_f16(aA, load_packed(pWg, g * 2 + 1, lane), d1);
    float b0 = bgp[g][m], b1 = bgp[g][16 + m];
#pragma unroll
    for (int v = 0; v < 8; ++v) {
      lds[(v + 8 * half) * 32 + m] = d0[v] + b0;
      lds[(v + 8 * half) * 32 + 16 + m] = d1[v] + b1;
    }
    __syncthreads();
    v16h a = load_a_f32(lds, 32, lane);
    __syncthreads();
    ((v16h*)outp[g])[tile * 32 + lane] = a;
  }
}

// Layer-2 GRU cell: GCN part already packed.
__global__ __launch_bounds__(32)
void l2_cell_kernel(const _Float16* __restrict__ gAz, const _Float16* __restrict__ gAr,
                    const _Float16* __restrict__ gAh, float* __restrict__ H,
                    float* __restrict__ acc,
                    const _Float16* __restrict__ pWl,  // [3][2][2]
                    const float* bl_z, const float* bl_r, const float* bl_h,
                    const float* probs, int t) {
  __shared__ float lds[16 * 32];
  const int tile = blockIdx.x, lane = threadIdx.x;
  const int m = lane & 15, half = lane >> 4;
  const int base = tile * 16;

  v16h aH = load_a_f32(H + (long)base * 32, 32, lane);
  v16h aGz = load_packed(gAz, tile, lane);
  v16h aGr = load_packed(gAr, tile, lane);
  v16h aGh = load_packed(gAh, tile, lane);

  v8f Hd0, Hd1;
#pragma unroll
  for (int v = 0; v < 8; ++v) {
    int row = base + v + 8 * half;
    Hd0[v] = H[row * 32 + m];
    Hd1[v] = H[row * 32 + 16 + m];
  }

  v8f zd0 = vzero8(), zd1 = vzero8();
  zd0 = wmma_f16(aGz, load_packed(pWl, 0 * 4 + 0, lane), zd0);
  zd0 = wmma_f16(aH, load_packed(pWl, 0 * 4 + 2, lane), zd0);
  zd1 = wmma_f16(aGz, load_packed(pWl, 0 * 4 + 1, lane), zd1);
  zd1 = wmma_f16(aH, load_packed(pWl, 0 * 4 + 3, lane), zd1);
  {
    float b0 = bl_z[m], b1 = bl_z[16 + m];
#pragma unroll
    for (int v = 0; v < 8; ++v) {
      zd0[v] = sigmoidf_(zd0[v] + b0);
      zd1[v] = sigmoidf_(zd1[v] + b1);
    }
  }
  v8f rd0 = vzero8(), rd1 = vzero8();
  rd0 = wmma_f16(aGr, load_packed(pWl, 1 * 4 + 0, lane), rd0);
  rd0 = wmma_f16(aH, load_packed(pWl, 1 * 4 + 2, lane), rd0);
  rd1 = wmma_f16(aGr, load_packed(pWl, 1 * 4 + 1, lane), rd1);
  rd1 = wmma_f16(aH, load_packed(pWl, 1 * 4 + 3, lane), rd1);
  {
    float b0 = bl_r[m], b1 = bl_r[16 + m];
#pragma unroll
    for (int v = 0; v < 8; ++v) {
      rd0[v] = sigmoidf_(rd0[v] + b0);
      rd1[v] = sigmoidf_(rd1[v] + b1);
    }
  }
#pragma unroll
  for (int v = 0; v < 8; ++v) {
    lds[(v + 8 * half) * 32 + m] = Hd0[v] * rd0[v];
    lds[(v + 8 * half) * 32 + 16 + m] = Hd1[v] * rd1[v];
  }
  __syncthreads();
  v16h aHR = load_a_f32(lds, 32, lane);
  __syncthreads();
  v8f hd0 = vzero8(), hd1 = vzero8();
  hd0 = wmma_f16(aGh, load_packed(pWl, 2 * 4 + 0, lane), hd0);
  hd0 = wmma_f16(aHR, load_packed(pWl, 2 * 4 + 2, lane), hd0);
  hd1 = wmma_f16(aGh, load_packed(pWl, 2 * 4 + 1, lane), hd1);
  hd1 = wmma_f16(aHR, load_packed(pWl, 2 * 4 + 3, lane), hd1);
  {
    float b0 = bl_h[m], b1 = bl_h[16 + m];
#pragma unroll
    for (int v = 0; v < 8; ++v) {
      hd0[v] = tanhf(hd0[v] + b0);
      hd1[v] = tanhf(hd1[v] + b1);
    }
  }
  float w = probs[t];
#pragma unroll
  for (int v = 0; v < 8; ++v) {
    int row = base + v + 8 * half;
    float hn0 = zd0[v] * Hd0[v] + (1.0f - zd0[v]) * hd0[v];
    float hn1 = zd1[v] * Hd1[v] + (1.0f - zd1[v]) * hd1[v];
    H[row * 32 + m] = hn0;
    H[row * 32 + 16 + m] = hn1;
    acc[row * 32 + m] += w * hn0;
    acc[row * 32 + 16 + m] += w * hn1;
  }
}

// ---------------------------------------------------------------- decoder

__global__ __launch_bounds__(32)
void decoder_kernel(const float* __restrict__ acc2,
                    const _Float16* __restrict__ pW1, const float* b1,
                    const _Float16* __restrict__ pW2, const float* b2,
                    const _Float16* __restrict__ pW3, const float* b3,
                    float* __restrict__ out) {
  __shared__ float lds[16 * 64];
  const int tile = blockIdx.x, lane = threadIdx.x;
  const int m = lane & 15, half = lane >> 4;
  const int base = tile * 16;

  // h = relu(acc2), A layout
  v16h aIn;
  {
    const float* r = acc2 + (long)base * 32 + m * 32;
#pragma unroll
    for (int e = 0; e < 8; ++e) aIn[e] = (_Float16)fmaxf(r[8 * half + e], 0.0f);
#pragma unroll
    for (int e = 0; e < 8; ++e) aIn[8 + e] = (_Float16)fmaxf(r[16 + 8 * half + e], 0.0f);
  }
  // y1 = relu(h @ W1 + b1), 16x64
#pragma unroll
  for (int nt = 0; nt < 4; ++nt) {
    v8f c = vzero8();
    c = wmma_f16(aIn, load_packed(pW1, nt, lane), c);
    float bb = b1[nt * 16 + m];
#pragma unroll
    for (int v = 0; v < 8; ++v)
      lds[(v + 8 * half) * 64 + nt * 16 + m] = fmaxf(c[v] + bb, 0.0f);
  }
  __syncthreads();
  v16h a1k0 = load_a_f32(lds, 64, lane);
  v16h a1k1 = load_a_f32(lds + 32, 64, lane);
  __syncthreads();
  // y2 = relu(y1 @ W2 + b2), 16x32
#pragma unroll
  for (int nt = 0; nt < 2; ++nt) {
    v8f c = vzero8();
    c = wmma_f16(a1k0, load_packed(pW2, 0 * 2 + nt, lane), c);
    c = wmma_f16(a1k1, load_packed(pW2, 1 * 2 + nt, lane), c);
    float bb = b2[nt * 16 + m];
#pragma unroll
    for (int v = 0; v < 8; ++v)
      lds[(v + 8 * half) * 32 + nt * 16 + m] = fmaxf(c[v] + bb, 0.0f);
  }
  __syncthreads();
  v16h a2 = load_a_f32(lds, 32, lane);
  __syncthreads();
  // y3 = y2 @ W3 + b3, 16x100 (112 padded)
#pragma unroll
  for (int nt = 0; nt < 7; ++nt) {
    v8f c = vzero8();
    c = wmma_f16(a2, load_packed(pW3, nt, lane), c);
    int col = nt * 16 + m;
    if (col < 100) {
      float bb = b3[col];
#pragma unroll
      for (int v = 0; v < 8; ++v) {
        int row = base + v + 8 * half;
        out[(long)row * 100 + col] = c[v] + bb;
      }
    }
  }
}

// ---------------------------------------------------------------- launcher

extern "C" void kernel_launch(void* const* d_in, const int* in_sizes, int n_in,
                              void* d_out, int out_size, void* d_ws, size_t ws_size,
                              hipStream_t stream) {
  const float* x = (const float*)d_in[0];
  const int* edge = (const int*)d_in[1];
  const int E = in_sizes[1] / 2;
  const int N = in_sizes[0] / (9 * 30);
  const int* src = edge;
  const int* dst = edge + E;

  const float* att1 = (const float*)d_in[2];
  const float* l1Wg[3] = {(const float*)d_in[3], (const float*)d_in[7], (const float*)d_in[11]};
  const float* l1bg[3] = {(const float*)d_in[4], (const float*)d_in[8], (const float*)d_in[12]};
  const float* l1Wl[3] = {(const float*)d_in[5], (const float*)d_in[9], (const float*)d_in[13]};
  const float* l1bl[3] = {(const float*)d_in[6], (const float*)d_in[10], (const float*)d_in[14]};
  const float* att2 = (const float*)d_in[15];
  const float* l2Wg[3] = {(const float*)d_in[16], (const float*)d_in[20], (const float*)d_in[24]};
  const float* l2bg[3] = {(const float*)d_in[17], (const float*)d_in[21], (const float*)d_in[25]};
  const float* l2Wl[3] = {(const float*)d_in[18], (const float*)d_in[22], (const float*)d_in[26]};
  const float* l2bl[3] = {(const float*)d_in[19], (const float*)d_in[23], (const float*)d_in[27]};
  const float* W1 = (const float*)d_in[28];
  const float* b1 = (const float*)d_in[29];
  const float* W2 = (const float*)d_in[30];
  const float* b2 = (const float*)d_in[31];
  const float* W3 = (const float*)d_in[32];
  const float* b3 = (const float*)d_in[33];
  float* out = (float*)d_out;

  // ---- workspace carve (256B aligned) ----
  char* w = (char*)d_ws;
  size_t off = 0;
  auto alloc = [&](size_t bytes) -> void* {
    void* p = w + off;
    off += (bytes + 255) & ~(size_t)255;
    return p;
  };
  float* probs1 = (float*)alloc(32 * 4);
  float* probs2 = (float*)alloc(32 * 4);
  float* deg = (float*)alloc((size_t)N * 4);
  float* dinv = (float*)alloc((size_t)N * 4);
  float* snorm = (float*)alloc((size_t)N * 4);
  float* enorm = (float*)alloc((size_t)E * 4);
  float* aggX = (float*)alloc((size_t)N * 16 * 4);
  float* H1 = (float*)alloc((size_t)N * 32 * 4);
  float* acc1 = (float*)alloc((size_t)N * 32 * 4);
  float* hL1 = (float*)alloc((size_t)N * 32 * 4);
  float* aggH = (float*)alloc((size_t)N * 32 * 4);
  float* H2 = (float*)alloc((size_t)N * 32 * 4);
  float* acc2 = (float*)alloc((size_t)N * 32 * 4);
  _Float16* gAz = (_Float16*)alloc((size_t)N * 32 * 2);
  _Float16* gAr = (_Float16*)alloc((size_t)N * 32 * 2);
  _Float16* gAh = (_Float16*)alloc((size_t)N * 32 * 2);
  _Float16* pWg1 = (_Float16*)alloc(3 * 2 * 512 * 2);
  _Float16* pWl1 = (_Float16*)alloc(3 * 4 * 512 * 2);
  _Float16* pWg2 = (_Float16*)alloc(3 * 2 * 512 * 2);
  _Float16* pWl2 = (_Float16*)alloc(3 * 4 * 512 * 2);
  _Float16* pW1 = (_Float16*)alloc(4 * 512 * 2);
  _Float16* pW2 = (_Float16*)alloc(4 * 512 * 2);
  _Float16* pW3 = (_Float16*)alloc(7 * 512 * 2);
  (void)ws_size; (void)n_in; (void)out_size;

  const int TPB = 256;
  const int tiles = N / 16;  // 3125, exact

  // graph norms
  fill_kernel<<<(N + TPB - 1) / TPB, TPB, 0, stream>>>(deg, 0.0f, N);
  deg_kernel<<<(E + TPB - 1) / TPB, TPB, 0, stream>>>(dst, deg, E);
  node_norm_kernel<<<(N + TPB - 1) / TPB, TPB, 0, stream>>>(deg, dinv, snorm, N);
  enorm_kernel<<<(E + TPB - 1) / TPB, TPB, 0, stream>>>(src, dst, dinv, enorm, E);

  softmax30_kernel<<<1, 32, 0, stream>>>(att1, probs1);
  softmax30_kernel<<<1, 32, 0, stream>>>(att2, probs2);

  // pack weights into WMMA B layout
  for (int g = 0; g < 3; ++g) {
    pack_b_kernel<<<2, 32, 0, stream>>>(l1Wg[g], 9, 32, 2, pWg1 + g * 2 * 512);
    pack_b_kernel<<<4, 32, 0, stream>>>(l1Wl[g], 64, 32, 2, pWl1 + g * 4 * 512);
    pack_b_kernel<<<2, 32, 0, stream>>>(l2Wg[g], 32, 32, 2, pWg2 + g * 2 * 512);
    pack_b_kernel<<<4, 32, 0, stream>>>(l2Wl[g], 64, 32, 2, pWl2 + g * 4 * 512);
  }
  pack_b_kernel<<<4, 32, 0, stream>>>(W1, 32, 64, 4, pW1);
  pack_b_kernel<<<4, 32, 0, stream>>>(W2, 64, 32, 2, pW2);
  pack_b_kernel<<<7, 32, 0, stream>>>(W3, 32, 100, 7, pW3);

  // ---- layer 1 temporal scan ----
  fill_kernel<<<((long)N * 32 + TPB - 1) / TPB, TPB, 0, stream>>>(H1, 0.0f, (long)N * 32);
  fill_kernel<<<((long)N * 32 + TPB - 1) / TPB, TPB, 0, stream>>>(acc1, 0.0f, (long)N * 32);
  for (int t = 0; t < 30; ++t) {
    aggx_init_kernel<<<(N * 16 + TPB - 1) / TPB, TPB, 0, stream>>>(x, snorm, aggX, t, N);
    aggx_edge_kernel<<<(E * 9 + TPB - 1) / TPB, TPB, 0, stream>>>(x, src, dst, enorm, aggX, t, E);
    l1_cell_kernel<<<tiles, 32, 0, stream>>>(aggX, H1, acc1, pWg1, pWl1,
                                             l1bg[0], l1bg[1], l1bg[2],
                                             l1bl[0], l1bl[1], l1bl[2], probs1, t);
  }
  relu_kernel<<<((long)N * 32 + TPB - 1) / TPB, TPB, 0, stream>>>(acc1, hL1, (long)N * 32);

  // ---- layer 2: aggregate once, precompute GCN outputs once ----
  aggh_init_kernel<<<(N * 32 + TPB - 1) / TPB, TPB, 0, stream>>>(hL1, snorm, aggH, N);
  aggh_edge_kernel<<<((long)E * 32 + TPB - 1) / TPB, TPB, 0, stream>>>(hL1, src, dst, enorm, aggH, E);
  l2_gcn_kernel<<<tiles, 32, 0, stream>>>(aggH, pWg2, l2bg[0], l2bg[1], l2bg[2],
                                          gAz, gAr, gAh);

  fill_kernel<<<((long)N * 32 + TPB - 1) / TPB, TPB, 0, stream>>>(H2, 0.0f, (long)N * 32);
  fill_kernel<<<((long)N * 32 + TPB - 1) / TPB, TPB, 0, stream>>>(acc2, 0.0f, (long)N * 32);
  for (int t = 0; t < 30; ++t) {
    l2_cell_kernel<<<tiles, 32, 0, stream>>>(gAz, gAr, gAh, H2, acc2, pWl2,
                                             l2bl[0], l2bl[1], l2bl[2], probs2, t);
  }

  // ---- decoder ----
  decoder_kernel<<<tiles, 32, 0, stream>>>(acc2, pW1, b1, pW2, b2, pW3, b3, out);
}
